// MultiHeadSelfAttention_4466765987956
// MI455X (gfx1250) — matrix operand order repaired and checked by reading, not compiled, not run on hardware
//
#include <hip/hip_runtime.h>

#define DEVINL __device__ __forceinline__

typedef __attribute__((ext_vector_type(16))) __bf16 v16bf;
typedef __attribute__((ext_vector_type(8)))  float  v8f;
typedef __attribute__((ext_vector_type(4)))  unsigned int u32x4;
typedef __attribute__((ext_vector_type(4)))  int i32x4;
typedef __attribute__((ext_vector_type(8)))  int i32x8;

constexpr int Bb   = 8;
constexpr int Ss   = 2048;
constexpr int DIN  = 768;
constexpr int Hh   = 3;
constexpr int DH   = 256;
constexpr int DOUT = 768;
constexpr int Mrows = Bb * Ss;        // 16384
constexpr float ATT_SCALE = 0.0625f;  // 1/sqrt(256)

#if defined(__has_builtin)
#if __has_builtin(__builtin_amdgcn_tensor_load_to_lds)
#define HAVE_TDM 1
#endif
#endif
#ifndef HAVE_TDM
#define HAVE_TDM 0
#endif

union FU { u32x4 h[2]; v16bf v; };

DEVINL unsigned short f2bf(float f) {
  unsigned int u = __float_as_uint(f);
  u += 0x7FFFu + ((u >> 16) & 1u);      // round-to-nearest-even truncate
  return (unsigned short)(u >> 16);
}
DEVINL unsigned int pack2(float lo, float hi) {
  return ((unsigned int)f2bf(hi) << 16) | (unsigned int)f2bf(lo);
}

DEVINL v8f wmma_bf16(v16bf a, v16bf b, v8f c) {
  return __builtin_amdgcn_wmma_f32_16x16x32_bf16(false, a, false, b, (short)0, c,
                                                 false, false);
}

// A-operand fragment: 16 rows x 32 k (bf16), source [row][k] row-major.
// lane&15 = row; lanes<16 hold k 0..15, lanes>=16 hold k 16..31 (contiguous).
DEVINL v16bf load_frag_a(const unsigned short* base, int rowStride, int lane) {
  FU f;
  const unsigned short* p = base + (lane & 15) * rowStride + ((lane < 16) ? 0 : 16);
  f.h[0] = *(const u32x4*)(p);
  f.h[1] = *(const u32x4*)(p + 8);
  return f.v;
}
// B-operand fragment: 32 k x 16 n (bf16), source [n][k] row-major (i.e. B^T).
// lane&15 = n; lanes<16 hold k 0..15, lanes>=16 hold k 16..31 (contiguous).
DEVINL v16bf load_frag_b(const unsigned short* base, int rowStride, int lane) {
  FU f;
  const unsigned short* p = base + (lane & 15) * rowStride + ((lane < 16) ? 0 : 16);
  f.h[0] = *(const u32x4*)(p);
  f.h[1] = *(const u32x4*)(p + 8);
  return f.v;
}

#if HAVE_TDM
// TDM DMA of a dense 2D bf16 tile [rows][row_elems] (row stride == row_elems
// elements in global memory) into LDS. Issued once per calling wave; tracked
// by TENSORcnt. D# layout per CDNA5 ISA §8 (group0 + group1; groups 2/3 null).
DEVINL void tdm_load_2d(const unsigned short* gsrc, unsigned short* ldst,
                        int rows, int row_elems) {
  unsigned long long ga = (unsigned long long)(uintptr_t)gsrc;
  unsigned lds = (unsigned)(uintptr_t)ldst;  // low 32 bits = LDS offset
  u32x4 g0 = {
      1u,                                  // count=1, user desc
      lds,                                 // lds_addr
      (unsigned)(ga & 0xFFFFFFFFu),        // global_addr[31:0]
      (unsigned)((ga >> 32) & 0x01FFFFFFu) // global_addr[56:32]
          | 0x80000000u                    // type=2 ("image")
  };
  unsigned dim0 = (unsigned)row_elems;     // tensor_dim0 (elements)
  unsigned dim1 = (unsigned)rows;          // tensor_dim1
  i32x8 g1;
  g1[0] = (int)(1u << 16);                 // wg_mask=0, data_size=1 (2 bytes)
  g1[1] = (int)((dim0 & 0xFFFFu) << 16);   // tensor_dim0[15:0] @ bit48
  g1[2] = (int)((dim0 >> 16) | ((dim1 & 0xFFFFu) << 16));  // dim0 hi, dim1 lo
  g1[3] = (int)((dim1 >> 16) | (dim0 << 16));              // dim1 hi, tile_dim0
  g1[4] = (int)(dim1 & 0xFFFFu);           // tile_dim1 (tile_dim2=0)
  g1[5] = (int)dim0;                       // tensor_dim0_stride[31:0]
  g1[6] = 0;                               // stride hi / dim1_stride lo
  g1[7] = 0;
  i32x4 z4 = {0, 0, 0, 0};
#if __clang_major__ >= 23
  i32x8 z8 = {0, 0, 0, 0, 0, 0, 0, 0};
  __builtin_amdgcn_tensor_load_to_lds(g0, g1, z4, z4, z8, 0);
#else
  __builtin_amdgcn_tensor_load_to_lds(g0, g1, z4, z4, 0);
#endif
}
#endif

// ---------------------------------------------------------------- converters
__global__ __launch_bounds__(256) void k_cvt_x(const float* __restrict__ x,
                                               unsigned short* __restrict__ xb) {
  int i = blockIdx.x * 256 + threadIdx.x;  // 4 floats per thread
  float4 f = ((const float4*)x)[i];
  uint2 o;
  o.x = pack2(f.x, f.y);
  o.y = pack2(f.z, f.w);
  ((uint2*)xb)[i] = o;
}

// wt[((mat*H + h)*DH + d)*DIN + i] = W_mat[h][i][d]   (bf16, transposed)
__global__ __launch_bounds__(256) void k_cvt_wqkv(const float* __restrict__ Wq,
                                                  const float* __restrict__ Wk,
                                                  const float* __restrict__ Wv,
                                                  unsigned short* __restrict__ wt) {
  int t = blockIdx.x * 256 + threadIdx.x;
  int i = t % DIN;
  int r = t / DIN;
  int d = r & 255;
  int r2 = r >> 8;
  int h = r2 % Hh;
  int mat = r2 / Hh;
  const float* W = (mat == 0) ? Wq : (mat == 1) ? Wk : Wv;
  wt[t] = f2bf(W[((size_t)h * DIN + i) * DH + d]);
}

// wpt[n*768 + k] = Wp[k*768 + n]
__global__ __launch_bounds__(256) void k_cvt_wp(const float* __restrict__ Wp,
                                                unsigned short* __restrict__ wpt) {
  int t = blockIdx.x * 256 + threadIdx.x;
  int n = t / (Hh * DH);
  int k = t % (Hh * DH);
  wpt[t] = f2bf(Wp[(size_t)k * DOUT + n]);
}

// ------------------------------------------------------------ fused QKV GEMM
// C[M=16384, N=2304] = xb * wt^T ; epilogue: +bias, q*=SCALE, scatter q/k/v
__global__ __launch_bounds__(256) void k_gemm_qkv(
    const unsigned short* __restrict__ xb, const unsigned short* __restrict__ wt,
    const float* __restrict__ bq, const float* __restrict__ bk,
    const float* __restrict__ bv, unsigned short* __restrict__ qb,
    unsigned short* __restrict__ kb, unsigned short* __restrict__ vb) {
  __shared__ unsigned short As[128 * 32];
  __shared__ unsigned short Bs[128 * 32];
  const int tid = threadIdx.x;
  const int lane = tid & 31;
  const int w = tid >> 5;
  const int wm = w & 3;   // 32-row slice
  const int wn = w >> 2;  // 64-col slice
  const int mbase = blockIdx.y * 128;
  const int nbase = blockIdx.x * 128;

  v8f acc[2][4];
#pragma unroll
  for (int i = 0; i < 2; ++i)
#pragma unroll
    for (int j = 0; j < 4; ++j)
#pragma unroll
      for (int g = 0; g < 8; ++g) acc[i][j][g] = 0.0f;

  for (int kk = 0; kk < DIN; kk += 32) {
#pragma unroll
    for (int it = 0; it < 2; ++it) {
      int u = tid + it * 256;
      int row = u >> 2;
      int c8 = (u & 3) * 8;
      *(u32x4*)&As[row * 32 + c8] =
          *(const u32x4*)&xb[(size_t)(mbase + row) * DIN + kk + c8];
      *(u32x4*)&Bs[row * 32 + c8] =
          *(const u32x4*)&wt[(size_t)(nbase + row) * DIN + kk + c8];
    }
    __syncthreads();
    if (kk + 32 < DIN)
      __builtin_prefetch(&xb[(size_t)(mbase + (tid >> 1)) * DIN + kk + 32], 0, 3);
    v16bf a0 = load_frag_a(&As[(wm * 32 + 0) * 32], 32, lane);
    v16bf a1 = load_frag_a(&As[(wm * 32 + 16) * 32], 32, lane);
#pragma unroll
    for (int j = 0; j < 4; ++j) {
      v16bf bfr = load_frag_b(&Bs[(wn * 64 + j * 16) * 32], 32, lane);
      acc[0][j] = wmma_bf16(a0, bfr, acc[0][j]);
      acc[1][j] = wmma_bf16(a1, bfr, acc[1][j]);
    }
    __syncthreads();
  }

  const int rowadd = (lane < 16) ? 0 : 8;
  const int col = lane & 15;
#pragma unroll
  for (int i = 0; i < 2; ++i) {
#pragma unroll
    for (int j = 0; j < 4; ++j) {
      int n = nbase + wn * 64 + j * 16 + col;
      int mat = n / 768;
      int rem = n - mat * 768;
      int h = rem >> 8;
      int d = rem & 255;
      const float* bias = (mat == 0) ? bq : (mat == 1) ? bk : bv;
      unsigned short* dst = (mat == 0) ? qb : (mat == 1) ? kb : vb;
      float bval = bias[h * 256 + d];
#pragma unroll
      for (int g = 0; g < 8; ++g) {
        int m = mbase + wm * 32 + i * 16 + g + rowadd;
        int bidx = m >> 11;  // /2048
        int s = m & 2047;
        float val = acc[i][j][g] + bval;
        if (mat == 0) val *= ATT_SCALE;  // fold softmax scale into q
        dst[(size_t)(((h * Bb + bidx) * Ss + s) << 8) + d] = f2bf(val);
      }
    }
  }
}

// ---------------------------------------------------------- flash attention
// grid: (S/128, B*H). 8 waves, 16 queries each, online softmax over 2048 keys.
// Double-buffered LDS ping-pong; K tiles DMA'd by the Tensor Data Mover
// (wave 0, TENSORcnt), V tiles register-prefetched + VALU-transposed.
__global__ __launch_bounds__(256) void k_attn(const unsigned short* __restrict__ qb,
                                              const unsigned short* __restrict__ kb,
                                              const unsigned short* __restrict__ vb,
                                              unsigned short* __restrict__ zb) {
  __shared__ unsigned short Ks[2][32 * 256];  // [key][d]
  __shared__ unsigned short Vt[2][256 * 32];  // [d][key]  (transposed)
  const int tid = threadIdx.x;
  const int lane = tid & 31;
  const int w = tid >> 5;
  const int b = blockIdx.y / Hh;
  const int h = blockIdx.y % Hh;
  const size_t base = (size_t)(h * Bb + b) * Ss * DH;
  const unsigned short* qp = qb + base;
  const unsigned short* kp = kb + base;
  const unsigned short* vp = vb + base;
  const int q0 = blockIdx.x * 128 + w * 16;
  const bool lohalf = (lane < 16);

  // Q resident in registers as B-operand fragments (16 rows x 256 features)
  v16bf qf[8];
#pragma unroll
  for (int c = 0; c < 8; ++c)
    qf[c] = load_frag_b(qp + (size_t)q0 * DH + c * 32, DH, lane);

  v8f o[16];
#pragma unroll
  for (int t = 0; t < 16; ++t)
#pragma unroll
    for (int g = 0; g < 8; ++g) o[t][g] = 0.0f;
  float mrow = -3.0e38f;
  float denom = 0.0f;

  // ---- stage tile 0 into buffer 0
  {
#if HAVE_TDM
    if (w == 0) tdm_load_2d(kp, &Ks[0][0], 32, DH);
#endif
#pragma unroll
    for (int it = 0; it < 4; ++it) {
      int u = tid + it * 256;
      int row = u >> 5;
      int c8 = (u & 31) * 8;
#if !HAVE_TDM
      *(u32x4*)&Ks[0][row * 256 + c8] =
          *(const u32x4*)&kp[(size_t)row * DH + c8];
#endif
      u32x4 val = *(const u32x4*)&vp[(size_t)row * DH + c8];
      const unsigned short* pv = (const unsigned short*)&val;
#pragma unroll
      for (int j = 0; j < 8; ++j) Vt[0][(c8 + j) * 32 + row] = pv[j];
    }
#if HAVE_TDM
    if (w == 0) __builtin_amdgcn_s_wait_tensorcnt(0);
#endif
    __syncthreads();
  }

  for (int kt = 0; kt < Ss; kt += 32) {
    const int buf = (kt >> 5) & 1;
    const int nbuf = buf ^ 1;
    const int nkt = kt + 32;
    const bool hasNext = (nkt < Ss);

    // issue next tile: K via TDM (async DMA), V into registers
    u32x4 vr[4];
#if !HAVE_TDM
    u32x4 kr[4];
#endif
    if (hasNext) {
#if HAVE_TDM
      if (w == 0) tdm_load_2d(kp + (size_t)nkt * DH, &Ks[nbuf][0], 32, DH);
#endif
#pragma unroll
      for (int it = 0; it < 4; ++it) {
        int u = tid + it * 256;
        int row = u >> 5;
        int c8 = (u & 31) * 8;
        vr[it] = *(const u32x4*)&vp[(size_t)(nkt + row) * DH + c8];
#if !HAVE_TDM
        kr[it] = *(const u32x4*)&kp[(size_t)(nkt + row) * DH + c8];
#endif
      }
    }

    // transposed scores: St[key][query] = K * Q^T  (scale pre-folded into q)
    v8f sc0, sc1;
#pragma unroll
    for (int g = 0; g < 8; ++g) { sc0[g] = 0.0f; sc1[g] = 0.0f; }
#pragma unroll
    for (int c = 0; c < 8; ++c) {
      v16bf ka0 = load_frag_a(&Ks[buf][0 * 256 + c * 32], 256, lane);
      v16bf ka1 = load_frag_a(&Ks[buf][16 * 256 + c * 32], 256, lane);
      sc0 = wmma_bf16(ka0, qf[c], sc0);
      sc1 = wmma_bf16(ka1, qf[c], sc1);
    }

    // online softmax: each lane holds 16 of its query's 32 tile-scores
    float lm = sc0[0];
#pragma unroll
    for (int g = 1; g < 8; ++g) lm = fmaxf(lm, sc0[g]);
#pragma unroll
    for (int g = 0; g < 8; ++g) lm = fmaxf(lm, sc1[g]);
    lm = fmaxf(lm, __shfl_xor(lm, 16, 32));
    float newm = fmaxf(mrow, lm);
    float r = __expf(mrow - newm);
    float p0[8], p1[8], ls = 0.0f;
#pragma unroll
    for (int g = 0; g < 8; ++g) { p0[g] = __expf(sc0[g] - newm); ls += p0[g]; }
#pragma unroll
    for (int g = 0; g < 8; ++g) { p1[g] = __expf(sc1[g] - newm); ls += p1[g]; }
    ls += __shfl_xor(ls, 16, 32);
    denom = denom * r + ls;
    mrow = newm;
#pragma unroll
    for (int t = 0; t < 16; ++t)
#pragma unroll
      for (int g = 0; g < 8; ++g) o[t][g] *= r;

    // P (C-layout of St) -> B-operand fragment: one cross-half shuffle + packs
    float e0[8], e1[8];
#pragma unroll
    for (int g = 0; g < 8; ++g) {
      e0[g] = __shfl_xor(p0[g], 16, 32);
      e1[g] = __shfl_xor(p1[g], 16, 32);
    }
    unsigned int uu[8];
#pragma unroll
    for (int rr = 0; rr < 4; ++rr) {
      float a0 = lohalf ? p0[2 * rr] : e1[2 * rr];
      float a1 = lohalf ? p0[2 * rr + 1] : e1[2 * rr + 1];
      uu[rr] = pack2(a0, a1);
      float c0 = lohalf ? e0[2 * rr] : p1[2 * rr];
      float c1 = lohalf ? e0[2 * rr + 1] : p1[2 * rr + 1];
      uu[rr + 4] = pack2(c0, c1);
    }
    FU pf;
    u32x4 lo4 = {uu[0], uu[1], uu[2], uu[3]};
    u32x4 hi4 = {uu[4], uu[5], uu[6], uu[7]};
    pf.h[0] = lo4;
    pf.h[1] = hi4;

    // O[d][query] += V^T[d][key] * P[key][query]
#pragma unroll
    for (int dt = 0; dt < 16; ++dt) {
      v16bf va = load_frag_a(&Vt[buf][dt * 16 * 32], 32, lane);
      o[dt] = wmma_bf16(va, pf.v, o[dt]);
    }

    // drain next-tile registers into the alternate buffers
    if (hasNext) {
#pragma unroll
      for (int it = 0; it < 4; ++it) {
        int u = tid + it * 256;
        int row = u >> 5;
        int c8 = (u & 31) * 8;
#if !HAVE_TDM
        *(u32x4*)&Ks[nbuf][row * 256 + c8] = kr[it];
#endif
        const unsigned short* pv = (const unsigned short*)&vr[it];
#pragma unroll
        for (int j = 0; j < 8; ++j) Vt[nbuf][(c8 + j) * 32 + row] = pv[j];
      }
    }
#if HAVE_TDM
    if (w == 0) __builtin_amdgcn_s_wait_tensorcnt(0);
#endif
    __syncthreads();
  }

  const float inv = 1.0f / denom;
  const int col = lane & 15;
  const int rowadd = lohalf ? 0 : 8;
  unsigned short* zr =
      zb + (size_t)(b * Ss + q0 + col) * (Hh * DH) + h * DH;
#pragma unroll
  for (int dt = 0; dt < 16; ++dt)
#pragma unroll
    for (int g = 0; g < 8; ++g)
      zr[dt * 16 + g + rowadd] = f2bf(o[dt][g] * inv);
}

// -------------------------------------------------------- output projection
__global__ __launch_bounds__(256) void k_gemm_proj(
    const unsigned short* __restrict__ zbuf, const unsigned short* __restrict__ wpt,
    const float* __restrict__ bp, float* __restrict__ out) {
  __shared__ unsigned short As[128 * 32];
  __shared__ unsigned short Bs[128 * 32];
  const int tid = threadIdx.x;
  const int lane = tid & 31;
  const int w = tid >> 5;
  const int wm = w & 3;
  const int wn = w >> 2;
  const int mbase = blockIdx.y * 128;
  const int nbase = blockIdx.x * 128;

  v8f acc[2][4];
#pragma unroll
  for (int i = 0; i < 2; ++i)
#pragma unroll
    for (int j = 0; j < 4; ++j)
#pragma unroll
      for (int g = 0; g < 8; ++g) acc[i][j][g] = 0.0f;

  for (int kk = 0; kk < Hh * DH; kk += 32) {
#pragma unroll
    for (int it = 0; it < 2; ++it) {
      int u = tid + it * 256;
      int row = u >> 2;
      int c8 = (u & 3) * 8;
      *(u32x4*)&As[row * 32 + c8] =
          *(const u32x4*)&zbuf[(size_t)(mbase + row) * (Hh * DH) + kk + c8];
      *(u32x4*)&Bs[row * 32 + c8] =
          *(const u32x4*)&wpt[(size_t)(nbase + row) * (Hh * DH) + kk + c8];
    }
    __syncthreads();
    v16bf a0 = load_frag_a(&As[(wm * 32 + 0) * 32], 32, lane);
    v16bf a1 = load_frag_a(&As[(wm * 32 + 16) * 32], 32, lane);
#pragma unroll
    for (int j = 0; j < 4; ++j) {
      v16bf bfr = load_frag_b(&Bs[(wn * 64 + j * 16) * 32], 32, lane);
      acc[0][j] = wmma_bf16(a0, bfr, acc[0][j]);
      acc[1][j] = wmma_bf16(a1, bfr, acc[1][j]);
    }
    __syncthreads();
  }

  const int rowadd = (lane < 16) ? 0 : 8;
  const int col = lane & 15;
#pragma unroll
  for (int i = 0; i < 2; ++i) {
#pragma unroll
    for (int j = 0; j < 4; ++j) {
      int n = nbase + wn * 64 + j * 16 + col;
      float bval = bp[n];
#pragma unroll
      for (int g = 0; g < 8; ++g) {
        int m = mbase + wm * 32 + i * 16 + g + rowadd;
        out[(size_t)m * DOUT + n] = acc[i][j][g] + bval;
      }
    }
  }
}

// -------------------------------------------------------------------- launch
extern "C" void kernel_launch(void* const* d_in, const int* in_sizes, int n_in,
                              void* d_out, int out_size, void* d_ws, size_t ws_size,
                              hipStream_t stream) {
  (void)in_sizes; (void)n_in; (void)out_size; (void)ws_size;
  const float* x  = (const float*)d_in[0];
  const float* Wq = (const float*)d_in[1];
  const float* bq = (const float*)d_in[2];
  const float* Wk = (const float*)d_in[3];
  const float* bk = (const float*)d_in[4];
  const float* Wv = (const float*)d_in[5];
  const float* bv = (const float*)d_in[6];
  const float* Wp = (const float*)d_in[7];
  const float* bp = (const float*)d_in[8];
  float* out = (float*)d_out;

  char* ws = (char*)d_ws;
  unsigned short* xb = (unsigned short*)ws;           // x in bf16 (aliased by z later)
  ws += (size_t)Mrows * DIN * 2;                      // 25_165_824 B
  unsigned short* wt = (unsigned short*)ws;           // 9 transposed QKV weights
  ws += (size_t)3 * Hh * DH * DIN * 2;                //  3_538_944 B
  unsigned short* qb = (unsigned short*)ws;
  ws += (size_t)Hh * Bb * Ss * DH * 2;                // 25_165_824 B
  unsigned short* kbuf = (unsigned short*)ws;
  ws += (size_t)Hh * Bb * Ss * DH * 2;
  unsigned short* vbuf = (unsigned short*)ws;
  ws += (size_t)Hh * Bb * Ss * DH * 2;
  unsigned short* wpt = (unsigned short*)ws;          //  1_179_648 B
  ws += (size_t)DOUT * Hh * DH * 2;
  unsigned short* zbuf = xb;  // xb dead after QKV GEMM; reuse for z

  k_cvt_x<<<Mrows * DIN / 1024, 256, 0, stream>>>(x, xb);
  k_cvt_wqkv<<<(3 * Hh * DH * DIN) / 256, 256, 0, stream>>>(Wq, Wk, Wv, wt);
  k_cvt_wp<<<(DOUT * Hh * DH) / 256, 256, 0, stream>>>(Wp, wpt);
  k_gemm_qkv<<<dim3(2304 / 128, Mrows / 128), 256, 0, stream>>>(
      xb, wt, bq, bk, bv, qb, kbuf, vbuf);
  k_attn<<<dim3(Ss / 128, Bb * Hh), 256, 0, stream>>>(qb, kbuf, vbuf, zbuf);
  k_gemm_proj<<<dim3(DOUT / 128, Mrows / 128), 256, 0, stream>>>(zbuf, wpt, bp, out);
}
